// EarlyRanker_23218593202349
// MI455X (gfx1250) — matrix operand order repaired and checked by reading, not compiled, not run on hardware
//
#include <hip/hip_runtime.h>
#include <hip/hip_bf16.h>

// ---------------------------------------------------------------------------
// CDNA5 (gfx1250) wave32 WMMA implementation of the EarlyRanker pipeline.
// All GEMMs run through v_wmma_f32_16x16x32_bf16 (bf16 operands, f32 acc).
// Register-blocked: one wave computes a 64x16 C strip (4 WMMA tiles) so the
// weight (B) operand is fetched once per K-step and reused 4x.
// ---------------------------------------------------------------------------

typedef __attribute__((ext_vector_type(16))) __bf16 v16bf;
typedef __attribute__((ext_vector_type(8)))  float  v8f;

union BfTile { unsigned u[8]; v16bf v; };

#define BATCH   16384
#define EMB_D   128
#define NSPARSE 26
#define NFEAT   27          // bm + 26 embeddings
#define TROW    (NFEAT * EMB_D)   // 3456
#define CMBW    480         // 128 + 351 triu + 1 pad col (=15*32)

// ---- WMMA operand loaders (ISA 7.12.2 16-bit layouts) ----------------------
// A (16x32, MxK): lanes 0-15 -> M=lane, K pairs {0..7,16..23}; lanes 16-31 ->
// M=lane-16, K pairs {8..15,24..31}.  K pairs are consecutive -> u32 loads.
__device__ __forceinline__ v16bf load_a16x32(const __bf16* base, int ldk,
                                             int row, int k0, int lane) {
  BfTile t;
  const unsigned* rp = (const unsigned*)(base + (long)row * ldk + k0);
  const int kb = (lane < 16) ? 0 : 8;
#pragma unroll
  for (int v = 0; v < 4; ++v) t.u[v] = rp[(kb + 2 * v) >> 1];
#pragma unroll
  for (int v = 4; v < 8; ++v) t.u[v] = rp[(16 + kb + 2 * (v - 4)) >> 1];
  return t.v;
}

// B (32x16, KxN): lane n (mod 16) holds column n; lanes 0-15: K {0..7,16..23},
// lanes 16-31: K {8..15,24..31}, two K per VGPR.  Column n of W^T == row n of W.
__device__ __forceinline__ v16bf load_b32x16(const __bf16* w, int ldk,
                                             int col, int k0, int lane) {
  BfTile t;
  const unsigned* rp = (const unsigned*)(w + (long)col * ldk + k0);
  const int kb = (lane < 16) ? 0 : 8;
#pragma unroll
  for (int v = 0; v < 4; ++v) t.u[v] = rp[(kb + 2 * v) >> 1];
#pragma unroll
  for (int v = 4; v < 8; ++v) t.u[v] = rp[(16 + kb + 2 * (v - 4)) >> 1];
  return t.v;
}

// ---- generic GEMM: C[M x N] = act(A[M x K] * W[N x K]^T + bias) ------------
// one wave per 64x16 C strip (4 WMMA tiles); 4 waves per block;
// grid = (N/16, M/256, batch)
__global__ void __launch_bounds__(128) gemm_bf16_kernel(
    const __bf16* __restrict__ A, int lda, long aBS,
    const __bf16* __restrict__ W, int K,  long wBS,
    const float*  __restrict__ bias, int biasBS,
    __bf16* __restrict__ C, int ldc, long cBS,
    int relu)
{
  const int lane = threadIdx.x & 31;
  const int wave = threadIdx.x >> 5;
  const int t    = blockIdx.z;
  const int n0   = blockIdx.x * 16;
  const int m0   = (blockIdx.y * 4 + wave) * 64;   // 4 M-tiles per wave

  const __bf16* a = A + (long)t * aBS;
  const __bf16* w = W + (long)t * wBS;
  __bf16*       c = C + (long)t * cBS;

  v8f acc0 = {}, acc1 = {}, acc2 = {}, acc3 = {};
  const int arl = lane & 15;
  const int bc  = n0 + (lane & 15);
  for (int k0 = 0; k0 < K; k0 += 32) {
    // pull next K-step of the streamed weight row ahead of time
    if (k0 + 32 < K)
      __builtin_prefetch((const void*)(w + (long)bc * K + k0 + 32), 0, 3);
    v16bf bv = load_b32x16(w, K, bc, k0, lane);
    v16bf a0 = load_a16x32(a, lda, m0 +  0 + arl, k0, lane);
    v16bf a1 = load_a16x32(a, lda, m0 + 16 + arl, k0, lane);
    v16bf a2 = load_a16x32(a, lda, m0 + 32 + arl, k0, lane);
    v16bf a3 = load_a16x32(a, lda, m0 + 48 + arl, k0, lane);
    acc0 = __builtin_amdgcn_wmma_f32_16x16x32_bf16(false, a0, false, bv, (short)0, acc0, false, false);
    acc1 = __builtin_amdgcn_wmma_f32_16x16x32_bf16(false, a1, false, bv, (short)0, acc1, false, false);
    acc2 = __builtin_amdgcn_wmma_f32_16x16x32_bf16(false, a2, false, bv, (short)0, acc2, false, false);
    acc3 = __builtin_amdgcn_wmma_f32_16x16x32_bf16(false, a3, false, bv, (short)0, acc3, false, false);
  }

  const float bn = bias ? bias[t * biasBS + n0 + (lane & 15)] : 0.f;
  const int mb = m0 + ((lane >> 4) << 3);   // VGPR r -> M = r (lo lanes) / r+8
  const int n  = n0 + (lane & 15);
  v8f accs[4] = {acc0, acc1, acc2, acc3};
#pragma unroll
  for (int i = 0; i < 4; ++i) {
#pragma unroll
    for (int r = 0; r < 8; ++r) {
      float v = accs[i][r] + bn;
      if (relu) v = v > 0.f ? v : 0.f;
      c[(long)(mb + i * 16 + r) * ldc + n] = (__bf16)v;
    }
  }
}

// ---- f32 -> bf16 conversion with per-row zero padding ----------------------
__global__ void convert_pad_kernel(const float* __restrict__ src,
                                   __bf16* __restrict__ dst,
                                   int rows, int sc, int dc) {
  long i = (long)blockIdx.x * blockDim.x + threadIdx.x;
  if (i >= (long)rows * dc) return;
  int r = (int)(i / dc), cidx = (int)(i % dc);
  float v = (cidx < sc) ? src[(long)r * sc + cidx] : 0.f;
  dst[i] = (__bf16)v;
}

// ---- embedding gather: one wave per (feature,batch) row --------------------
__global__ void __launch_bounds__(256) gather_kernel(
    const int* __restrict__ emb_idx, const float* __restrict__ tables,
    __bf16* __restrict__ T)
{
  const int lane = threadIdx.x & 31;
  const int rid  = blockIdx.x * 8 + (threadIdx.x >> 5);  // f*B + b
  const int f = rid >> 14;
  const int b = rid & (BATCH - 1);
  const int idx = emb_idx[f * BATCH + b];
  const float4* src = (const float4*)(tables + ((long)f * 100000 + idx) * EMB_D);
  __bf16* dst = T + ((long)b * NFEAT + 1 + f) * EMB_D;
  float4 v = src[lane];
  __bf16 p[4] = {(__bf16)v.x, (__bf16)v.y, (__bf16)v.z, (__bf16)v.w};
  *(uint2*)(dst + lane * 4) = *(const uint2*)p;
}

// ---- per-sample interaction Z = T T^T, scatter triu(k=1) -------------------
__device__ __forceinline__ void scatter_triu(const v8f& c, int tm, int tn,
                                             int lane, __bf16* dst) {
  const int n  = tn + (lane & 15);
  const int mb = tm + ((lane >> 4) << 3);
#pragma unroll
  for (int r = 0; r < 8; ++r) {
    int m = mb + r;
    if (m < NFEAT && n < NFEAT && m < n) {
      int idx = 26 * m - (m * (m - 1)) / 2 + (n - m - 1);
      dst[EMB_D + idx] = (__bf16)c[r];
    }
  }
}

__global__ void __launch_bounds__(256) interact_kernel(
    const __bf16* __restrict__ T, __bf16* __restrict__ combined)
{
  const int lane = threadIdx.x & 31;
  const int wave = threadIdx.x >> 5;
  const long b = (long)blockIdx.x * 8 + wave;
  const __bf16* tb = T + b * TROW;

  v8f c00 = {}, c01 = {}, c10 = {}, c11 = {};
  const int r0 = lane & 15;
  const int rHi = (16 + r0 < NFEAT) ? 16 + r0 : NFEAT - 1;  // clamp OOB rows
  for (int k0 = 0; k0 < EMB_D; k0 += 32) {
    v16bf a0 = load_a16x32(tb, EMB_D, r0,  k0, lane);
    v16bf a1 = load_a16x32(tb, EMB_D, rHi, k0, lane);
    v16bf b0 = load_b32x16(tb, EMB_D, r0,  k0, lane);
    v16bf b1 = load_b32x16(tb, EMB_D, rHi, k0, lane);
    c00 = __builtin_amdgcn_wmma_f32_16x16x32_bf16(false, a0, false, b0, (short)0, c00, false, false);
    c01 = __builtin_amdgcn_wmma_f32_16x16x32_bf16(false, a0, false, b1, (short)0, c01, false, false);
    c10 = __builtin_amdgcn_wmma_f32_16x16x32_bf16(false, a1, false, b0, (short)0, c10, false, false);
    c11 = __builtin_amdgcn_wmma_f32_16x16x32_bf16(false, a1, false, b1, (short)0, c11, false, false);
  }

  __bf16* cmb = combined + b * CMBW;
  for (int j = lane; j < EMB_D; j += 32) cmb[j] = tb[j];  // bm -> cols 0..127
  if (lane == 0) cmb[CMBW - 1] = (__bf16)0.f;             // zero pad col 479
  scatter_triu(c00,  0,  0, lane, cmb);
  scatter_triu(c01,  0, 16, lane, cmb);
  scatter_triu(c10, 16,  0, lane, cmb);
  scatter_triu(c11, 16, 16, lane, cmb);
}

// ---- final per-task dot (N=1 output) ---------------------------------------
__global__ void tower_out_kernel(const __bf16* __restrict__ t1,
                                 const float* __restrict__ tw2,
                                 const float* __restrict__ tb2,
                                 float* __restrict__ out)
{
  int i = blockIdx.x * blockDim.x + threadIdx.x;      // t*B + b
  int t = i >> 14;
  int b = i & (BATCH - 1);
  const __bf16* x = t1 + ((long)t * BATCH + b) * 64;
  const float*  w = tw2 + t * 64;
  float s = tb2[t];
#pragma unroll 8
  for (int j = 0; j < 64; ++j) s += (float)x[j] * w[j];
  out[i] = s;
}

// ---------------------------------------------------------------------------
extern "C" void kernel_launch(void* const* d_in, const int* in_sizes, int n_in,
                              void* d_out, int out_size, void* d_ws, size_t ws_size,
                              hipStream_t stream) {
  const float* dense   = (const float*)d_in[0];
  const int*   emb_idx = (const int*)  d_in[1];
  const float* emb_tab = (const float*)d_in[2];
  const float* bw0 = (const float*)d_in[3];  const float* bb0 = (const float*)d_in[4];
  const float* bw1 = (const float*)d_in[5];  const float* bb1 = (const float*)d_in[6];
  const float* bw2 = (const float*)d_in[7];  const float* bb2 = (const float*)d_in[8];
  const float* pw  = (const float*)d_in[9];  const float* pb  = (const float*)d_in[10];
  const float* sw0 = (const float*)d_in[11]; const float* sb0 = (const float*)d_in[12];
  const float* sw1 = (const float*)d_in[13]; const float* sb1 = (const float*)d_in[14];
  const float* tw0 = (const float*)d_in[15]; const float* tb0 = (const float*)d_in[16];
  const float* tw1 = (const float*)d_in[17]; const float* tb1 = (const float*)d_in[18];
  const float* tw2 = (const float*)d_in[19]; const float* tb2 = (const float*)d_in[20];
  float* out = (float*)d_out;

  char* ws = (char*)d_ws;
  size_t o = 0;
  auto alloc = [&](size_t bytes) { size_t r = o; o += (bytes + 255) & ~(size_t)255; return r; };

  __bf16* densePad = (__bf16*)(ws + alloc((size_t)BATCH * 32 * 2));
  __bf16* bw0b = (__bf16*)(ws + alloc((size_t)512 * 32  * 2));
  __bf16* bw1b = (__bf16*)(ws + alloc((size_t)256 * 512 * 2));
  __bf16* bw2b = (__bf16*)(ws + alloc((size_t)128 * 256 * 2));
  __bf16* pwb  = (__bf16*)(ws + alloc((size_t)256 * CMBW * 2));
  __bf16* sw0b = (__bf16*)(ws + alloc((size_t)512 * 256 * 2));
  __bf16* sw1b = (__bf16*)(ws + alloc((size_t)256 * 512 * 2));
  __bf16* tw0b = (__bf16*)(ws + alloc((size_t)4 * 128 * 256 * 2));
  __bf16* tw1b = (__bf16*)(ws + alloc((size_t)4 * 64 * 128 * 2));
  __bf16* h0   = (__bf16*)(ws + alloc((size_t)BATCH * 512 * 2));   // reused as s1
  __bf16* h1   = (__bf16*)(ws + alloc((size_t)BATCH * 256 * 2));   // reused as user
  __bf16* Tb   = (__bf16*)(ws + alloc((size_t)BATCH * TROW * 2));  // later carved
  __bf16* cmb  = (__bf16*)(ws + alloc((size_t)BATCH * CMBW * 2));
  // Carved out of Tb after interaction stage (T is dead by then):
  __bf16* sbuf = Tb;
  __bf16* t0b  = (__bf16*)((char*)Tb + (size_t)BATCH * 256 * 2);
  __bf16* t1b  = (__bf16*)((char*)t0b + (size_t)4 * BATCH * 128 * 2);
  __bf16* s1   = h0;
  __bf16* user = h1;

  auto cvt = [&](const float* s, __bf16* d, int rows, int sc, int dc) {
    long n = (long)rows * dc;
    convert_pad_kernel<<<dim3((unsigned)((n + 255) / 256)), dim3(256), 0, stream>>>(s, d, rows, sc, dc);
  };

  // 0) weight / dense conversion (f32 -> bf16, K padded to mult of 32)
  cvt(dense, densePad, BATCH, 13, 32);
  cvt(bw0,  bw0b, 512, 13, 32);
  cvt(bw1,  bw1b, 256, 512, 512);
  cvt(bw2,  bw2b, 128, 256, 256);
  cvt(pw,   pwb,  256, 479, CMBW);
  cvt(sw0,  sw0b, 512, 256, 256);
  cvt(sw1,  sw1b, 256, 512, 512);
  cvt(tw0,  tw0b, 4 * 128, 256, 256);
  cvt(tw1,  tw1b, 4 * 64, 128, 128);

  const dim3 blk(128);
  const int my = BATCH / 256;   // 64  (4 waves x 64 rows per block)

  // 1) bottom MLP
  gemm_bf16_kernel<<<dim3(512 / 16, my, 1), blk, 0, stream>>>(
      densePad, 32, 0, bw0b, 32, 0, bb0, 0, h0, 512, 0, 1);
  gemm_bf16_kernel<<<dim3(256 / 16, my, 1), blk, 0, stream>>>(
      h0, 512, 0, bw1b, 512, 0, bb1, 0, h1, 256, 0, 1);
  // bm written directly into T row 0 (ldc = 3456)
  gemm_bf16_kernel<<<dim3(128 / 16, my, 1), blk, 0, stream>>>(
      h1, 256, 0, bw2b, 256, 0, bb2, 0, Tb, TROW, 0, 0);

  // 2) embedding gather into T rows 1..26
  gather_kernel<<<dim3((NSPARSE * BATCH) / 8), dim3(256), 0, stream>>>(
      emb_idx, emb_tab, Tb);

  // 3) pairwise interaction -> combined [B x 480]
  interact_kernel<<<dim3(BATCH / 8), dim3(256), 0, stream>>>(Tb, cmb);

  // 4) user / shared MLP
  gemm_bf16_kernel<<<dim3(256 / 16, my, 1), blk, 0, stream>>>(
      cmb, CMBW, 0, pwb, CMBW, 0, pb, 0, user, 256, 0, 0);
  gemm_bf16_kernel<<<dim3(512 / 16, my, 1), blk, 0, stream>>>(
      user, 256, 0, sw0b, 256, 0, sb0, 0, s1, 512, 0, 1);
  gemm_bf16_kernel<<<dim3(256 / 16, my, 1), blk, 0, stream>>>(
      s1, 512, 0, sw1b, 512, 0, sb1, 0, sbuf, 256, 0, 0);

  // 5) task towers (batched over z = 4 tasks)
  gemm_bf16_kernel<<<dim3(128 / 16, my, 4), blk, 0, stream>>>(
      sbuf, 256, 0, tw0b, 256, (long)128 * 256, tb0, 128,
      t0b, 128, (long)BATCH * 128, 1);
  gemm_bf16_kernel<<<dim3(64 / 16, my, 4), blk, 0, stream>>>(
      t0b, 128, (long)BATCH * 128, tw1b, 128, (long)64 * 128, tb1, 64,
      t1b, 64, (long)BATCH * 64, 1);

  // 6) final per-task scalar head -> out[4*B]
  tower_out_kernel<<<dim3((4 * BATCH) / 256), dim3(256), 0, stream>>>(
      t1b, tw2, tb2, out);
}
